// GMMLinear_90898687853038
// MI455X (gfx1250) — matrix operand order repaired and checked by reading, not compiled, not run on hardware
//
#include <hip/hip_runtime.h>
#include <hip/hip_bf16.h>
#include <stdint.h>

#define IN_F   1024
#define OUT_F  1024
#define NGMM   5
#define NROWS  ((IN_F + 1) * OUT_F)   /* 1,049,600 */
#define BATCH  4096

typedef __attribute__((ext_vector_type(16))) __bf16 v16bf;
typedef __attribute__((ext_vector_type(8)))  __bf16 v8bf;
typedef __attribute__((ext_vector_type(4)))  __bf16 v4bf;
typedef __attribute__((ext_vector_type(8)))  float  v8f;
typedef __attribute__((ext_vector_type(4)))  float  v4f;

// ---------------------------------------------------------------------------
// Threefry-2x32 (JAX round structure: 5 groups of 4 rounds, key injection)
// ---------------------------------------------------------------------------
__host__ __device__ inline uint32_t rotl32(uint32_t v, int d) {
  return (v << d) | (v >> (32 - d));
}

__host__ __device__ inline void threefry2x32(uint32_t k0, uint32_t k1,
                                             uint32_t x0, uint32_t x1,
                                             uint32_t& o0, uint32_t& o1) {
  uint32_t ks2 = k0 ^ k1 ^ 0x1BD11BDAu;
  x0 += k0; x1 += k1;
#define TF_R(r) { x0 += x1; x1 = rotl32(x1, (r)); x1 ^= x0; }
  TF_R(13) TF_R(15) TF_R(26) TF_R(6)   x0 += k1;  x1 += ks2 + 1u;
  TF_R(17) TF_R(29) TF_R(16) TF_R(24)  x0 += ks2; x1 += k0  + 2u;
  TF_R(13) TF_R(15) TF_R(26) TF_R(6)   x0 += k0;  x1 += k1  + 3u;
  TF_R(17) TF_R(29) TF_R(16) TF_R(24)  x0 += k1;  x1 += ks2 + 4u;
  TF_R(13) TF_R(15) TF_R(26) TF_R(6)   x0 += ks2; x1 += k0  + 5u;
#undef TF_R
  o0 = x0; o1 = x1;
}

// random_bits(n even): element i pairs with i +/- n/2; i<n/2 takes o0 else o1
__device__ inline uint32_t rand_bits_at(uint32_t k0, uint32_t k1,
                                        uint32_t i, uint32_t half_n) {
  uint32_t o0, o1;
  if (i < half_n) { threefry2x32(k0, k1, i, i + half_n, o0, o1); return o0; }
  threefry2x32(k0, k1, i - half_n, i, o0, o1); return o1;
}

__device__ inline float bits_to_unit(uint32_t bits) {  // [0,1)
  uint32_t u = (bits >> 9) | 0x3f800000u;
  return __uint_as_float(u) - 1.0f;
}

// ---------------------------------------------------------------------------
// Kernel 1: x f32 -> bf16
// ---------------------------------------------------------------------------
__global__ void gmm_cvt_x(const float* __restrict__ x, __bf16* __restrict__ xb,
                          int n4) {
  int i = blockIdx.x * blockDim.x + threadIdx.x;
  if (i >= n4) return;
  v4f v = ((const v4f*)x)[i];
  v4bf o;
  o.x = (__bf16)v.x; o.y = (__bf16)v.y; o.z = (__bf16)v.z; o.w = (__bf16)v.w;
  ((v4bf*)xb)[i] = o;
}

// ---------------------------------------------------------------------------
// Kernel 2: sample weight_bias rows (Gumbel-max categorical + reparam normal)
// ---------------------------------------------------------------------------
__global__ __launch_bounds__(256)
void gmm_sample(const float* __restrict__ mean,
                const float* __restrict__ logstd,
                const float* __restrict__ wlogit,
                float* __restrict__ bias,
                __bf16* __restrict__ Wb,
                uint32_t kc0, uint32_t kc1, uint32_t kn0, uint32_t kn1) {
  uint32_t r = blockIdx.x * blockDim.x + threadIdx.x;
  if (r >= (uint32_t)NROWS) return;

  const uint32_t HC = (uint32_t)NROWS * NGMM / 2u;  // 2,624,000
  const uint32_t HN = (uint32_t)NROWS / 2u;         //   524,800
  const float    TINY = 1.17549435e-38f;            // finfo(f32).tiny

  // categorical via Gumbel-max (first-max tie rule like jnp.argmax)
  int best = 0; float bestz = -3.402823466e38f;
#pragma unroll
  for (int g = 0; g < NGMM; ++g) {
    uint32_t i = r * NGMM + (uint32_t)g;
    float f = bits_to_unit(rand_bits_at(kc0, kc1, i, HC));
    float u = fmaxf(TINY, f * (1.0f - TINY) + TINY);
    float z = -logf(-logf(u)) + wlogit[i];
    if (z > bestz) { bestz = z; best = g; }
  }

  float m  = mean  [r * NGMM + best];
  float ls = logstd[r * NGMM + best];
  float sd = expf(tanhf(ls) * 3.0f);   // MAX_STD = 3

  // normal eps: uniform in [nextafter(-1,0), 1), sqrt(2)*erfinv
  const float LO = -0.99999994f;
  float f = bits_to_unit(rand_bits_at(kn0, kn1, r, HN));
  float u = fmaxf(LO, f * (1.0f - LO) + LO);
  float eps = 1.41421356237309505f * erfinvf(u);

  float wb = m + sd * eps;
  if (r < OUT_F) bias[r] = wb;
  else           Wb[r - OUT_F] = (__bf16)wb;
}

// ---------------------------------------------------------------------------
// Kernel 3: out = Xb(4096x1024) * Wb^T(1024x1024) + bias, f32 out
// Block: 256 thr = 8 waves; block tile 128Mx128N; wave tile 32Mx64N.
// Per-lane fragment layouts follow CDNA5 ISA tables (16-bit A 16x32, B 32x16).
// ---------------------------------------------------------------------------
__global__ __launch_bounds__(256)
void gmm_wmma_gemm(const __bf16* __restrict__ X, const __bf16* __restrict__ Wb,
                   const float* __restrict__ bias, float* __restrict__ out) {
  const int lane  = threadIdx.x & 31;
  const int wave  = threadIdx.x >> 5;     // 0..7
  const int waveM = wave & 3;             // 0..3 -> 4 waves in M
  const int waveN = wave >> 2;            // 0..1 -> 2 waves in N
  const int m_base = blockIdx.x * 128 + waveM * 32;
  const int n_base = blockIdx.y * 128 + waveN * 64;
  const int lr = lane & 15;
  const int hi = lane >> 4;               // lane half

  v8f acc[2][4];
#pragma unroll
  for (int i = 0; i < 2; ++i)
#pragma unroll
    for (int j = 0; j < 4; ++j) acc[i][j] = (v8f){};

  // A lane base: row m, K offset hi*8   (lane<16: K 0..7 & 16..23)
  const __bf16* Arow[2];
#pragma unroll
  for (int i = 0; i < 2; ++i)
    Arow[i] = X + (size_t)(m_base + i * 16 + lr) * IN_F + hi * 8;
  // B lane base: col n (row of W), K offset hi*8 (lane<16: K 0..7 & 16..23)
  const __bf16* Brow[4];
#pragma unroll
  for (int j = 0; j < 4; ++j)
    Brow[j] = Wb + (size_t)(n_base + j * 16 + lr) * IN_F + hi * 8;

  for (int k0 = 0; k0 < IN_F; k0 += 32) {
    v16bf a[2], b[4];
#pragma unroll
    for (int i = 0; i < 2; ++i) {
      union { v16bf v; v8bf h[2]; } u;
      u.h[0] = *(const v8bf*)(Arow[i] + k0);        // K = k0+hi*8 .. +7
      u.h[1] = *(const v8bf*)(Arow[i] + k0 + 16);   // K = k0+16+hi*8 .. +7
      a[i] = u.v;
    }
#pragma unroll
    for (int j = 0; j < 4; ++j) {
      union { v16bf v; v8bf h[2]; } u;
      u.h[0] = *(const v8bf*)(Brow[j] + k0);        // K = k0+hi*8 .. +7
      u.h[1] = *(const v8bf*)(Brow[j] + k0 + 16);   // K = k0+16+hi*8 .. +7
      b[j] = u.v;
    }
#pragma unroll
    for (int i = 0; i < 2; ++i)
#pragma unroll
      for (int j = 0; j < 4; ++j)
        acc[i][j] = __builtin_amdgcn_wmma_f32_16x16x32_bf16(
            false, a[i], false, b[j], (short)0, acc[i][j], false, false);
  }

  // D layout: VGPR r -> M = r (lanes 0-15) / r+8 (lanes 16-31), N = lane%16
#pragma unroll
  for (int i = 0; i < 2; ++i) {
    const int mrow = m_base + i * 16 + hi * 8;
#pragma unroll
    for (int j = 0; j < 4; ++j) {
      const int col = n_base + j * 16 + lr;
      const float bs = bias[col];
#pragma unroll
      for (int r = 0; r < 8; ++r)
        out[(size_t)(mrow + r) * OUT_F + col] = acc[i][j][r] + bs;
    }
  }
}

// ---------------------------------------------------------------------------
extern "C" void kernel_launch(void* const* d_in, const int* in_sizes, int n_in,
                              void* d_out, int out_size, void* d_ws, size_t ws_size,
                              hipStream_t stream) {
  const float* x      = (const float*)d_in[0];
  const float* mean   = (const float*)d_in[1];
  const float* logstd = (const float*)d_in[2];
  const float* wlogit = (const float*)d_in[3];
  float* outp = (float*)d_out;

  // workspace layout: W bf16 (2 MB) | bias f32 (4 KB) | x bf16 (8 MB)
  char* ws = (char*)d_ws;
  __bf16* Wb   = (__bf16*)ws;
  float*  bias = (float*)(ws + (size_t)OUT_F * IN_F * 2);
  __bf16* Xb   = (__bf16*)(ws + (size_t)OUT_F * IN_F * 2 + (size_t)OUT_F * 4);

  // host-side (deterministic, capture-safe) split of jax.random.key(42):
  // counts iota(4) -> pairs (0,2),(1,3); kcat=(o0a,o0b), knorm=(o1a,o1b)
  uint32_t a0, a1, b0, b1;
  threefry2x32(0u, 42u, 0u, 2u, a0, a1);
  threefry2x32(0u, 42u, 1u, 3u, b0, b1);

  const int n4 = BATCH * IN_F / 4;
  gmm_cvt_x<<<(n4 + 255) / 256, 256, 0, stream>>>(x, Xb, n4);

  gmm_sample<<<(NROWS + 255) / 256, 256, 0, stream>>>(
      mean, logstd, wlogit, bias, Wb, a0, b0, a1, b1);

  dim3 grid(BATCH / 128, OUT_F / 128);  // 32 x 8
  gmm_wmma_gemm<<<grid, 256, 0, stream>>>(Xb, Wb, bias, outp);
}
